// FeatureAggregator_45861660787026
// MI455X (gfx1250) — matrix operand order repaired and checked by reading, not compiled, not run on hardware
//
#include <hip/hip_runtime.h>

typedef __attribute__((ext_vector_type(16))) _Float16 v16h;
typedef __attribute__((ext_vector_type(8)))  float    v8f;
typedef __attribute__((ext_vector_type(8)))  _Float16 h8v;

#define TT  1024          // sequence length
#define TPD 1040          // padded row stride: 8 halo + 1024 + 8 halo (2080B, 16B-mult)
#define LDA 72            // LDS row stride in halfs (144B: 16B aligned, bank-friendly)

// ---------------------------------------------------------------------------
// WMMA fragment helpers (CDNA5 16x16x32 f16, wave32 layouts per ISA 7.12.2)
// ---------------------------------------------------------------------------
__device__ __forceinline__ v16h frag_a(const _Float16* As, int m0, int k0, int lane) {
  // A is [m][k] row-major in LDS, leading dim LDA.
  v16h a;
  const int row  = lane & 15;
  const int base = (lane >> 4) << 3;
  const _Float16* p = As + (size_t)(m0 + row) * LDA + k0;
#pragma unroll
  for (int i = 0; i < 8; ++i) a[i] = p[base + i];
#pragma unroll
  for (int i = 0; i < 8; ++i) a[8 + i] = p[16 + base + i];
  return a;
}

__device__ __forceinline__ v16h frag_b(const _Float16* Bs, int n0, int k0, int lane) {
  // B stored transposed in LDS: [n][k], leading dim LDA.
  v16h b;
  const _Float16* p = Bs + (size_t)(n0 + (lane & 15)) * LDA + k0 + ((lane >> 4) << 3);
#pragma unroll
  for (int i = 0; i < 8; ++i) b[i] = p[i];
#pragma unroll
  for (int i = 0; i < 8; ++i) b[8 + i] = p[16 + i];
  return b;
}

__device__ __forceinline__ v8f wmma16(v16h a, v16h b, v8f c) {
  return __builtin_amdgcn_wmma_f32_16x16x32_f16(false, a, false, b, (short)0, c, false, false);
}

// ---------------------------------------------------------------------------
// CDNA5 async global->LDS copies (VGLOBAL encoding; ASYNCcnt).
// INST_OFFSET is added to BOTH the LDS and global addresses (ISA 15.18.3),
// so consecutive offsets implement a contiguous per-lane copy.
// LDS aperture flat pointers carry the LDS byte address in their low 32 bits.
// ---------------------------------------------------------------------------
__device__ __forceinline__ void async_copy64(const _Float16* g, _Float16* l) {
  unsigned lo = (unsigned)(size_t)l;
  asm volatile(
      "global_load_async_to_lds_b128 %0, %1, off\n\t"
      "global_load_async_to_lds_b128 %0, %1, off offset:16\n\t"
      "global_load_async_to_lds_b128 %0, %1, off offset:32\n\t"
      "global_load_async_to_lds_b128 %0, %1, off offset:48"
      :: "v"(lo), "v"(g) : "memory");
}
__device__ __forceinline__ void async_copy32(const _Float16* g, _Float16* l) {
  unsigned lo = (unsigned)(size_t)l;
  asm volatile(
      "global_load_async_to_lds_b128 %0, %1, off\n\t"
      "global_load_async_to_lds_b128 %0, %1, off offset:16"
      :: "v"(lo), "v"(g) : "memory");
}
__device__ __forceinline__ void async_wait() {
  asm volatile("s_wait_asynccnt 0x0" ::: "memory");
}

// ---------------------------------------------------------------------------
// Layout/convert kernels
// ---------------------------------------------------------------------------
__global__ __launch_bounds__(256) void x_to_ct_kernel(const float* __restrict__ x,
                                                      _Float16* __restrict__ xh,
                                                      int Bn, int C) {
  size_t total = (size_t)Bn * TT * C;
  for (size_t i = (size_t)blockIdx.x * 256 + threadIdx.x; i < total;
       i += (size_t)gridDim.x * 256) {
    int c = (int)(i % C);
    size_t bt = i / C;
    int t = (int)(bt % TT);
    int b = (int)(bt / TT);
    xh[((size_t)b * C + c) * TPD + 8 + t] = (_Float16)x[i];
  }
}

// zero the 8+8 halo columns of each padded row
__global__ __launch_bounds__(256) void zero_halo_kernel(_Float16* __restrict__ buf, int rows) {
  size_t total = (size_t)rows * 16;
  for (size_t i = (size_t)blockIdx.x * 256 + threadIdx.x; i < total;
       i += (size_t)gridDim.x * 256) {
    int j = (int)(i % 16);
    size_t r = i / 16;
    buf[r * TPD + (j < 8 ? j : TT + j)] = (_Float16)0.f;
  }
}

// (Cout, Cin, K) f32  ->  (K, Cout, Cin) f16   (K==1 is an identity copy)
__global__ __launch_bounds__(256) void wconv_kernel(const float* __restrict__ w,
                                                    _Float16* __restrict__ wh,
                                                    int Cout, int Cin, int K) {
  size_t total = (size_t)Cout * Cin * K;
  for (size_t i = (size_t)blockIdx.x * 256 + threadIdx.x; i < total;
       i += (size_t)gridDim.x * 256) {
    int k = (int)(i % K);
    size_t r = i / K;
    int ci = (int)(r % Cin);
    int co = (int)(r / Cin);
    wh[((size_t)k * Cout + co) * Cin + ci] = (_Float16)w[i];
  }
}

// ---------------------------------------------------------------------------
// Conv-as-GEMM: Y[b,co,t] = epi( sum_tap sum_ci W[tap][co][ci]*X[b,ci,t+off(tap)] )
// Block tile 128(M) x 64(N), K tile 64, 8 waves (4x2), wave tile 32x32.
// Extended B tile (96 cols) loaded once per K-chunk; taps = shifted LDS reads.
// A tiles copied global->LDS with CDNA5 async-to-LDS (no VGPR staging).
// ---------------------------------------------------------------------------
__global__ __launch_bounds__(256)
void gemm_conv_kernel(const _Float16* __restrict__ W, const _Float16* __restrict__ X,
                      _Float16* __restrict__ Y,
                      const float* __restrict__ bias, const float* __restrict__ bnp,
                      const float* __restrict__ wscale, int widx,
                      int Cout, int Cin,
                      int ntaps, int off0, int off1, int off2, int off3, int off4,
                      int relu, int accum, int outCtot, int outCoff)
{
  __shared__ _Float16 As[128 * LDA];
  __shared__ _Float16 Bs[96 * LDA];
  const int offs[5] = {off0, off1, off2, off3, off4};
  const int tid  = threadIdx.x;
  const int lane = tid & 31;
  const int wave = tid >> 5;
  const int wm   = wave & 3;
  const int wn   = wave >> 2;
  const int m_blk = blockIdx.y * 128;
  const int n_blk = blockIdx.x * 64;
  const int b      = n_blk / TT;
  const int t_base = n_blk % TT;

  v8f acc[2][2];
#pragma unroll
  for (int mi = 0; mi < 2; ++mi)
#pragma unroll
    for (int ni = 0; ni < 2; ++ni)
#pragma unroll
      for (int r = 0; r < 8; ++r) acc[mi][ni][r] = 0.f;

  const _Float16* Xb = X + (size_t)b * Cin * TPD;
  const int a_row = tid >> 1;
  const int a_col = (tid & 1) * 32;
  const int b_kk  = tid >> 2;
  const int b_nb  = (tid & 3) * 24;
  _Float16* a_dst = &As[a_row * LDA + a_col];

  for (int k0 = 0; k0 < Cin; k0 += 64) {
    __syncthreads();
    // A tile for tap 0: async DMA global->LDS (64B per thread)
    async_copy64(W + (size_t)(m_blk + a_row) * Cin + k0 + a_col, a_dst);
    { // extended B tile: 96(n: t_base-8 .. t_base+87) x 64(k), transposing scatter
      const h8v* src = (const h8v*)(Xb + (size_t)(k0 + b_kk) * TPD + t_base + b_nb);
      h8v r0 = src[0], r1 = src[1], r2 = src[2];
#pragma unroll
      for (int i = 0; i < 8; ++i) Bs[(b_nb + i)      * LDA + b_kk] = r0[i];
#pragma unroll
      for (int i = 0; i < 8; ++i) Bs[(b_nb + 8 + i)  * LDA + b_kk] = r1[i];
#pragma unroll
      for (int i = 0; i < 8; ++i) Bs[(b_nb + 16 + i) * LDA + b_kk] = r2[i];
    }
    for (int tap = 0; tap < ntaps; ++tap) {
      if (tap) {
        __syncthreads();   // previous tap's compute done before overwriting As
        async_copy64(W + (size_t)tap * Cout * Cin +
                     (size_t)(m_blk + a_row) * Cin + k0 + a_col, a_dst);
      }
      async_wait();
      __syncthreads();
      const int shift = 8 + offs[tap];
#pragma unroll
      for (int ks = 0; ks < 64; ks += 32) {
        v16h af0 = frag_a(As, wm * 32,      ks, lane);
        v16h af1 = frag_a(As, wm * 32 + 16, ks, lane);
        v16h bf0 = frag_b(Bs, wn * 32 + shift,      ks, lane);
        v16h bf1 = frag_b(Bs, wn * 32 + 16 + shift, ks, lane);
        acc[0][0] = wmma16(af0, bf0, acc[0][0]);
        acc[0][1] = wmma16(af0, bf1, acc[0][1]);
        acc[1][0] = wmma16(af1, bf0, acc[1][0]);
        acc[1][1] = wmma16(af1, bf1, acc[1][1]);
      }
    }
  }

  const float alpha = wscale ? wscale[widx] : 1.f;
  const int hi8 = (lane >> 4) * 8;
#pragma unroll
  for (int mi = 0; mi < 2; ++mi) {
#pragma unroll
    for (int ni = 0; ni < 2; ++ni) {
      const int gn = n_blk + wn * 32 + ni * 16 + (lane & 15);
      const int t  = gn % TT;
      const int bb = gn / TT;
#pragma unroll
      for (int r = 0; r < 8; ++r) {
        const int gm = m_blk + wm * 32 + mi * 16 + hi8 + r;
        float v = acc[mi][ni][r];
        if (bias) v += bias[gm];
        if (relu) v = fmaxf(v, 0.f);
        if (bnp) {
          const float g  = bnp[gm];
          const float be = bnp[Cout + gm];
          const float mu = bnp[2 * Cout + gm];
          const float va = bnp[3 * Cout + gm];
          v = g * (v - mu) * rsqrtf(va + 1e-5f) + be;
        }
        v *= alpha;
        const size_t oidx = ((size_t)bb * outCtot + outCoff + gm) * TPD + 8 + t;
        if (accum) v += (float)Y[oidx];
        Y[oidx] = (_Float16)v;
      }
    }
  }
}

// ---------------------------------------------------------------------------
// Attention scores: S[h][q][k] = 0.125 * sum_d Q[h*64+d][q] * K[512+h*64+d][k]
// Per-batch launch; grid = (16, 8, H). Kdim = 64 (single tile).
// ---------------------------------------------------------------------------
__global__ __launch_bounds__(256)
void attn_scores_kernel(const _Float16* __restrict__ QKVb, float* __restrict__ S)
{
  __shared__ _Float16 As[128 * LDA];
  __shared__ _Float16 Bs[64 * LDA];
  const int h = blockIdx.z;
  const _Float16* Q  = QKVb + (size_t)(h * 64) * TPD + 8;
  const _Float16* Kp = QKVb + (size_t)(512 + h * 64) * TPD + 8;
  const int tid = threadIdx.x, lane = tid & 31, wave = tid >> 5;
  const int wm = wave & 3, wn = wave >> 2;
  const int m_blk = blockIdx.y * 128;
  const int n_blk = blockIdx.x * 64;

  { // A[m=q][k=d]: contiguous in q, transposing scatter into LDS
    const int d = tid >> 2, mb = (tid & 3) * 32;
    const h8v* src = (const h8v*)(Q + (size_t)d * TPD + m_blk + mb);
    h8v r0 = src[0], r1 = src[1], r2 = src[2], r3 = src[3];
#pragma unroll
    for (int i = 0; i < 8; ++i) As[(mb + i)      * LDA + d] = r0[i];
#pragma unroll
    for (int i = 0; i < 8; ++i) As[(mb + 8 + i)  * LDA + d] = r1[i];
#pragma unroll
    for (int i = 0; i < 8; ++i) As[(mb + 16 + i) * LDA + d] = r2[i];
#pragma unroll
    for (int i = 0; i < 8; ++i) As[(mb + 24 + i) * LDA + d] = r3[i];
  }
  { // B[k=d][n]: stored [n][k]
    const int d = tid >> 2, nb = (tid & 3) * 16;
    const h8v* src = (const h8v*)(Kp + (size_t)d * TPD + n_blk + nb);
    h8v r0 = src[0], r1 = src[1];
#pragma unroll
    for (int i = 0; i < 8; ++i) Bs[(nb + i)     * LDA + d] = r0[i];
#pragma unroll
    for (int i = 0; i < 8; ++i) Bs[(nb + 8 + i) * LDA + d] = r1[i];
  }
  __syncthreads();

  v8f acc[2][2];
#pragma unroll
  for (int mi = 0; mi < 2; ++mi)
#pragma unroll
    for (int ni = 0; ni < 2; ++ni)
#pragma unroll
      for (int r = 0; r < 8; ++r) acc[mi][ni][r] = 0.f;

#pragma unroll
  for (int ks = 0; ks < 64; ks += 32) {
    v16h af0 = frag_a(As, wm * 32,      ks, lane);
    v16h af1 = frag_a(As, wm * 32 + 16, ks, lane);
    v16h bf0 = frag_b(Bs, wn * 32,      ks, lane);
    v16h bf1 = frag_b(Bs, wn * 32 + 16, ks, lane);
    acc[0][0] = wmma16(af0, bf0, acc[0][0]);
    acc[0][1] = wmma16(af0, bf1, acc[0][1]);
    acc[1][0] = wmma16(af1, bf0, acc[1][0]);
    acc[1][1] = wmma16(af1, bf1, acc[1][1]);
  }

  const int hi8 = (lane >> 4) * 8;
#pragma unroll
  for (int mi = 0; mi < 2; ++mi)
#pragma unroll
    for (int ni = 0; ni < 2; ++ni) {
      const int gn = n_blk + wn * 32 + ni * 16 + (lane & 15);
#pragma unroll
      for (int r = 0; r < 8; ++r) {
        const int gm = m_blk + wm * 32 + mi * 16 + hi8 + r;
        S[((size_t)h * TT + gm) * TT + gn] = acc[mi][ni][r] * 0.125f;
      }
    }
}

// Row softmax over 1024 f32 -> f16. One wave32 per row, 8 rows per block.
__global__ __launch_bounds__(256)
void softmax_kernel(const float* __restrict__ S, _Float16* __restrict__ P)
{
  const int lane = threadIdx.x & 31;
  const size_t row = (size_t)blockIdx.x * 8 + (threadIdx.x >> 5);
  const float* s = S + row * TT;
  _Float16* p = P + row * TT;
  float vals[32];
  float mx = -1e30f;
#pragma unroll
  for (int j = 0; j < 32; ++j) { vals[j] = s[j * 32 + lane]; mx = fmaxf(mx, vals[j]); }
  for (int m = 16; m > 0; m >>= 1) mx = fmaxf(mx, __shfl_xor(mx, m, 32));
  float sum = 0.f;
#pragma unroll
  for (int j = 0; j < 32; ++j) { vals[j] = __expf(vals[j] - mx); sum += vals[j]; }
  for (int m = 16; m > 0; m >>= 1) sum += __shfl_xor(sum, m, 32);
  const float inv = 1.f / sum;
#pragma unroll
  for (int j = 0; j < 32; ++j) p[j * 32 + lane] = (_Float16)(vals[j] * inv);
}

// O[h*64+d][q] = sum_k P[h][q][k] * V[1024+h*64+d][k]
// Block tile 64(M=d) x 128(N=q), waves 2x4, Kdim = 1024 loop. Per-batch launch.
// Both tiles are contiguous copies -> async direct-to-LDS.
__global__ __launch_bounds__(256)
void attn_av_kernel(const _Float16* __restrict__ QKVb, const _Float16* __restrict__ P,
                    _Float16* __restrict__ Ab)
{
  __shared__ _Float16 As[64 * LDA];
  __shared__ _Float16 Bs[128 * LDA];
  const int h = blockIdx.z;
  const _Float16* V  = QKVb + (size_t)(1024 + h * 64) * TPD + 8;
  const _Float16* Ph = P + (size_t)h * TT * TT;
  const int tid = threadIdx.x, lane = tid & 31, wave = tid >> 5;
  const int wm = wave & 1, wn = wave >> 1;
  const int n_blk = blockIdx.x * 128;

  const int a_row = tid >> 2, a_cb = (tid & 3) * 16;
  const int b_n   = tid >> 1, b_kb = (tid & 1) * 32;
  _Float16* a_dst = &As[a_row * LDA + a_cb];
  _Float16* b_dst = &Bs[b_n * LDA + b_kb];

  v8f acc[2][2];
#pragma unroll
  for (int mi = 0; mi < 2; ++mi)
#pragma unroll
    for (int ni = 0; ni < 2; ++ni)
#pragma unroll
      for (int r = 0; r < 8; ++r) acc[mi][ni][r] = 0.f;

  for (int k0 = 0; k0 < TT; k0 += 64) {
    __syncthreads();
    async_copy32(V + (size_t)a_row * TPD + k0 + a_cb, a_dst);
    async_copy64(Ph + (size_t)(n_blk + b_n) * TT + k0 + b_kb, b_dst);
    async_wait();
    __syncthreads();
#pragma unroll
    for (int ks = 0; ks < 64; ks += 32) {
      v16h af0 = frag_a(As, wm * 32,      ks, lane);
      v16h af1 = frag_a(As, wm * 32 + 16, ks, lane);
      v16h bf0 = frag_b(Bs, wn * 32,      ks, lane);
      v16h bf1 = frag_b(Bs, wn * 32 + 16, ks, lane);
      acc[0][0] = wmma16(af0, bf0, acc[0][0]);
      acc[0][1] = wmma16(af0, bf1, acc[0][1]);
      acc[1][0] = wmma16(af1, bf0, acc[1][0]);
      acc[1][1] = wmma16(af1, bf1, acc[1][1]);
    }
  }

  const int hi8 = (lane >> 4) * 8;
#pragma unroll
  for (int mi = 0; mi < 2; ++mi)
#pragma unroll
    for (int ni = 0; ni < 2; ++ni) {
      const int gq = n_blk + wn * 32 + ni * 16 + (lane & 15);
#pragma unroll
      for (int r = 0; r < 8; ++r) {
        const int gm = wm * 32 + mi * 16 + hi8 + r;   // d in 0..63
        Ab[(size_t)(h * 64 + gm) * TPD + 8 + gq] = (_Float16)acc[mi][ni][r];
      }
    }
}

// LayerNorm over C=512 channels of (Xres + Delta), padded (B,C,TPD) layout.
__global__ __launch_bounds__(256)
void ln_add_kernel(const _Float16* __restrict__ Xr, const _Float16* __restrict__ Dl,
                   const float* __restrict__ g, const float* __restrict__ be,
                   _Float16* __restrict__ O)
{
  const int C = 512;
  const int lane = threadIdx.x & 31;
  const size_t tok = (size_t)blockIdx.x * 8 + (threadIdx.x >> 5);
  const int b = (int)(tok / TT), t = (int)(tok % TT);
  const size_t base = ((size_t)b * C) * TPD + 8 + t;
  float v[16];
  float mean = 0.f;
#pragma unroll
  for (int j = 0; j < 16; ++j) {
    const int c = j * 32 + lane;
    const float x = (float)Xr[base + (size_t)c * TPD] + (float)Dl[base + (size_t)c * TPD];
    v[j] = x; mean += x;
  }
  for (int m = 16; m > 0; m >>= 1) mean += __shfl_xor(mean, m, 32);
  mean *= (1.f / 512.f);
  float var = 0.f;
#pragma unroll
  for (int j = 0; j < 16; ++j) { const float d = v[j] - mean; var += d * d; }
  for (int m = 16; m > 0; m >>= 1) var += __shfl_xor(var, m, 32);
  const float rs = rsqrtf(var * (1.f / 512.f) + 1e-5f);
#pragma unroll
  for (int j = 0; j < 16; ++j) {
    const int c = j * 32 + lane;
    O[base + (size_t)c * TPD] = (_Float16)((v[j] - mean) * rs * g[c] + be[c]);
  }
}

// out[b][t][c] = relu(bb + feat)(b,c,t) + x[b][t][c]
__global__ __launch_bounds__(256)
void final_kernel(const _Float16* __restrict__ bb, const _Float16* __restrict__ feat,
                  const float* __restrict__ x, float* __restrict__ out,
                  int Bn, int C)
{
  size_t total = (size_t)Bn * TT * C;
  for (size_t i = (size_t)blockIdx.x * 256 + threadIdx.x; i < total;
       i += (size_t)gridDim.x * 256) {
    int c = (int)(i % C);
    size_t bt = i / C;
    int t = (int)(bt % TT);
    int b = (int)(bt / TT);
    const size_t ci = ((size_t)b * C + c) * TPD + 8 + t;
    const float v = (float)bb[ci] + (float)feat[ci];
    out[i] = fmaxf(v, 0.f) + x[i];
  }
}

// ---------------------------------------------------------------------------
// Host orchestration
// ---------------------------------------------------------------------------
static void launch_gemm(hipStream_t s, const _Float16* W, const _Float16* X, _Float16* Y,
                        const float* bias, const float* bnp, const float* wsc, int widx,
                        int Cout, int Cin, int ntaps,
                        int o0, int o1, int o2, int o3, int o4,
                        int relu, int accum, int outCtot, int outCoff)
{
  dim3 g(8 * TT / 64, Cout / 128);
  gemm_conv_kernel<<<g, 256, 0, s>>>(W, X, Y, bias, bnp, wsc, widx, Cout, Cin,
                                     ntaps, o0, o1, o2, o3, o4, relu, accum, outCtot, outCoff);
}

extern "C" void kernel_launch(void* const* d_in, const int* in_sizes, int n_in,
                              void* d_out, int out_size, void* d_ws, size_t ws_size,
                              hipStream_t stream) {
  (void)in_sizes; (void)n_in; (void)out_size; (void)ws_size;
  const int B = 8, D = 2048, L = 2;

  const float* x       = (const float*)d_in[0];
  const float* wts     = (const float*)d_in[1];
  const float* cw[4]   = {(const float*)d_in[2], (const float*)d_in[5],
                          (const float*)d_in[8], (const float*)d_in[11]};
  const float* cb[4]   = {(const float*)d_in[3], (const float*)d_in[6],
                          (const float*)d_in[9], (const float*)d_in[12]};
  const float* bnb[4]  = {(const float*)d_in[4], (const float*)d_in[7],
                          (const float*)d_in[10], (const float*)d_in[13]};
  const float* cw5     = (const float*)d_in[14];
  const float* tcw_in  = (const float*)d_in[15];
  const float* tcb_in  = (const float*)d_in[16];
  const float* t_in_w  = (const float*)d_in[17];
  const float* t_in_b  = (const float*)d_in[18];
  const float* t_out_w = (const float*)d_in[19];
  const float* t_out_b = (const float*)d_in[20];
  const float* ln1_g   = (const float*)d_in[21];
  const float* ln1_b   = (const float*)d_in[22];
  const float* ff_w1   = (const float*)d_in[23];
  const float* ff_b1   = (const float*)d_in[24];
  const float* ff_w2   = (const float*)d_in[25];
  const float* ff_b2   = (const float*)d_in[26];
  const float* ln2_g   = (const float*)d_in[27];
  const float* ln2_b   = (const float*)d_in[28];
  const float* tcw_out = (const float*)d_in[29];
  const float* tcb_out = (const float*)d_in[30];
  const float* cw6     = (const float*)d_in[31];
  const float* bn6     = (const float*)d_in[32];
  const float* bw1     = (const float*)d_in[33];
  const float* bb1     = (const float*)d_in[34];
  const float* bw2     = (const float*)d_in[35];
  const float* bb2     = (const float*)d_in[36];
  const float* bw3     = (const float*)d_in[37];
  const float* bb3     = (const float*)d_in[38];

  // ---- workspace bump allocator ----
  char* p = (char*)d_ws;
  auto ah = [&](size_t n) { _Float16* r = (_Float16*)p;
                            p += ((n * sizeof(_Float16) + 255) & ~(size_t)255); return r; };
  auto af = [&](size_t n) { float* r = (float*)p;
                            p += ((n * sizeof(float) + 255) & ~(size_t)255); return r; };

  _Float16* Xh     = ah((size_t)B * D * TPD);        // (B,C,TPD) f16; reused as bbuf
  _Float16* ff2048 = ah((size_t)B * 2048 * TPD);     // FFN hidden; reused as feat
  _Float16* cat    = ah((size_t)B * 1024 * TPD);     // concat [out_d | trans_out]
  _Float16* a512A  = ah((size_t)B * 512 * TPD);
  _Float16* a512B  = ah((size_t)B * 512 * TPD);
  _Float16* curA   = ah((size_t)B * 512 * TPD);
  _Float16* curB   = ah((size_t)B * 512 * TPD);
  _Float16* qkv    = ah((size_t)B * 1536 * TPD);
  _Float16* attn   = ah((size_t)B * 512 * TPD);
  _Float16* proj   = ah((size_t)B * 512 * TPD);
  float*    Sbuf   = af((size_t)8 * TT * TT);        // per-batch scores (H,T,T)
  _Float16* Pbuf   = ah((size_t)8 * TT * TT);        // per-batch softmax (H,T,T)
  // converted weights (K,Cout,Cin) f16
  _Float16* cwh[4]; cwh[0] = ah((size_t)3*512*D); cwh[1] = ah((size_t)3*512*D);
  cwh[2] = ah((size_t)3*512*D); cwh[3] = ah((size_t)5*512*D);
  _Float16* cw5h    = ah((size_t)512 * D);
  _Float16* tcwinh  = ah((size_t)512 * 512);
  _Float16* tinwh   = ah((size_t)L * 1536 * 512);
  _Float16* toutwh  = ah((size_t)L * 512 * 512);
  _Float16* ffw1h   = ah((size_t)L * 2048 * 512);
  _Float16* ffw2h   = ah((size_t)L * 512 * 2048);
  _Float16* tcwouth = ah((size_t)512 * 512);
  _Float16* cw6h    = ah((size_t)3 * 2048 * 1024);
  _Float16* bw1h    = ah((size_t)512 * 2048);
  _Float16* bw2h    = ah((size_t)3 * 512 * 512);
  _Float16* bw3h    = ah((size_t)2048 * 512);
  (void)ah(256);                                     // slack for ext-tile over-read
  _Float16* bbuf = Xh;       // alias: Xh dead after conv5
  _Float16* feat = ff2048;   // alias: FFN hidden dead after transformer

  // ---- stage 0: layout, halos, weight conversion ----
  x_to_ct_kernel<<<4096, 256, 0, stream>>>(x, Xh, B, D);
  zero_halo_kernel<<<256, 256, 0, stream>>>(Xh,    B * D);
  zero_halo_kernel<<<256, 256, 0, stream>>>(cat,   B * 1024);
  zero_halo_kernel<<<256, 256, 0, stream>>>(a512A, B * 512);
  wconv_kernel<<<2048, 256, 0, stream>>>(cw[0], cwh[0], 512, D, 3);
  wconv_kernel<<<2048, 256, 0, stream>>>(cw[1], cwh[1], 512, D, 3);
  wconv_kernel<<<2048, 256, 0, stream>>>(cw[2], cwh[2], 512, D, 3);
  wconv_kernel<<<2048, 256, 0, stream>>>(cw[3], cwh[3], 512, D, 5);
  wconv_kernel<<<2048, 256, 0, stream>>>(cw5, cw5h, 512, D, 1);
  wconv_kernel<<<512,  256, 0, stream>>>(tcw_in, tcwinh, 512, 512, 1);
  wconv_kernel<<<2048, 256, 0, stream>>>(t_in_w, tinwh, L * 1536, 512, 1);
  wconv_kernel<<<1024, 256, 0, stream>>>(t_out_w, toutwh, L * 512, 512, 1);
  wconv_kernel<<<2048, 256, 0, stream>>>(ff_w1, ffw1h, L * 2048, 512, 1);
  wconv_kernel<<<2048, 256, 0, stream>>>(ff_w2, ffw2h, L * 512, 2048, 1);
  wconv_kernel<<<512,  256, 0, stream>>>(tcw_out, tcwouth, 512, 512, 1);
  wconv_kernel<<<2048, 256, 0, stream>>>(cw6, cw6h, 2048, 1024, 3);
  wconv_kernel<<<1024, 256, 0, stream>>>(bw1, bw1h, 512, 2048, 1);
  wconv_kernel<<<1024, 256, 0, stream>>>(bw2, bw2h, 512, 512, 3);
  wconv_kernel<<<1024, 256, 0, stream>>>(bw3, bw3h, 2048, 512, 1);

  // ---- stage 1: four dilated conv branches -> cat[:,0:512,:] (weighted sum) ----
  launch_gemm(stream, cwh[0], Xh, cat, cb[0], bnb[0], wts, 0, 512, D, 3, -1,0,1,0,0, 1, 0, 1024, 0);
  launch_gemm(stream, cwh[1], Xh, cat, cb[1], bnb[1], wts, 1, 512, D, 3, -2,0,2,0,0, 1, 1, 1024, 0);
  launch_gemm(stream, cwh[2], Xh, cat, cb[2], bnb[2], wts, 2, 512, D, 3, -4,0,4,0,0, 1, 1, 1024, 0);
  launch_gemm(stream, cwh[3], Xh, cat, cb[3], bnb[3], wts, 3, 512, D, 5, -8,-4,0,4,8, 1, 1, 1024, 0);

  // ---- stage 2: conv5 (ReLU, no bias) then transformer input 1x1 conv ----
  launch_gemm(stream, cw5h, Xh, a512A, nullptr, nullptr, nullptr, 0, 512, D, 1, 0,0,0,0,0, 1, 0, 512, 0);
  launch_gemm(stream, tcwinh, a512A, curA, tcb_in, nullptr, nullptr, 0, 512, 512, 1, 0,0,0,0,0, 0, 0, 512, 0);

  // ---- stage 3: transformer layers ----
  _Float16* cur = curA; _Float16* nxt = curB;
  for (int i = 0; i < L; ++i) {
    launch_gemm(stream, tinwh + (size_t)i * 1536 * 512, cur, qkv,
                t_in_b + (size_t)i * 1536, nullptr, nullptr, 0,
                1536, 512, 1, 0,0,0,0,0, 0, 0, 1536, 0);
    for (int b = 0; b < B; ++b) {
      const _Float16* qb = qkv + (size_t)b * 1536 * TPD;
      attn_scores_kernel<<<dim3(16, 8, 8), 256, 0, stream>>>(qb, Sbuf);
      softmax_kernel<<<(8 * TT) / 8, 256, 0, stream>>>(Sbuf, Pbuf);
      attn_av_kernel<<<dim3(8, 1, 8), 256, 0, stream>>>(qb, Pbuf, attn + (size_t)b * 512 * TPD);
    }
    launch_gemm(stream, toutwh + (size_t)i * 512 * 512, attn, proj,
                t_out_b + (size_t)i * 512, nullptr, nullptr, 0,
                512, 512, 1, 0,0,0,0,0, 0, 0, 512, 0);
    ln_add_kernel<<<(B * TT) / 8, 256, 0, stream>>>(cur, proj, ln1_g + (size_t)i * 512,
                                                    ln1_b + (size_t)i * 512, nxt);
    { _Float16* tmp = cur; cur = nxt; nxt = tmp; }
    launch_gemm(stream, ffw1h + (size_t)i * 2048 * 512, cur, ff2048,
                ff_b1 + (size_t)i * 2048, nullptr, nullptr, 0,
                2048, 512, 1, 0,0,0,0,0, 1, 0, 2048, 0);
    launch_gemm(stream, ffw2h + (size_t)i * 512 * 2048, ff2048, proj,
                ff_b2 + (size_t)i * 512, nullptr, nullptr, 0,
                512, 2048, 1, 0,0,0,0,0, 0, 0, 512, 0);
    ln_add_kernel<<<(B * TT) / 8, 256, 0, stream>>>(cur, proj, ln2_g + (size_t)i * 512,
                                                    ln2_b + (size_t)i * 512, nxt);
    { _Float16* tmp = cur; cur = nxt; nxt = tmp; }
  }

  // ---- stage 4: transformer output conv -> cat[:,512:1024,:] ----
  launch_gemm(stream, tcwouth, cur, cat, tcb_out, nullptr, nullptr, 0,
              512, 512, 1, 0,0,0,0,0, 0, 0, 1024, 512);

  // ---- stage 5: cw6 (k=3, ReLU, BN) -> feat ----
  launch_gemm(stream, cw6h, cat, feat, nullptr, bn6, nullptr, 0,
              2048, 1024, 3, -1,0,1,0,0, 1, 0, 2048, 0);

  // ---- stage 6: bottleneck ----
  launch_gemm(stream, bw1h, feat,  a512A, bb1, nullptr, nullptr, 0, 512, 2048, 1, 0,0,0,0,0, 1, 0, 512, 0);
  launch_gemm(stream, bw2h, a512A, a512B, bb2, nullptr, nullptr, 0, 512, 512, 3, -1,0,1,0,0, 1, 0, 512, 0);
  launch_gemm(stream, bw3h, a512B, bbuf,  bb3, nullptr, nullptr, 0, 2048, 512, 1, 0,0,0,0,0, 0, 0, 2048, 0);

  // ---- stage 7: relu(b + feat) + residual(x), transpose to (B,T,2048) f32 ----
  final_kernel<<<8192, 256, 0, stream>>>(bbuf, feat, x, (float*)d_out, B, D);
}
